// EncoderMoe_24223615549897
// MI455X (gfx1250) — compile-verified, matrix operand order checked by reading, not run on hardware
//
#include <hip/hip_runtime.h>
#include <hip/hip_bf16.h>

// ---------------------------------------------------------------------------
// Config (matches reference)
// ---------------------------------------------------------------------------
#define NUM_EXPERTS 8
#define D_MODEL 768
#define MLP_DIM 3072
// det: B=16,S=1024 -> 16 groups of 1024, C=256, rows/expert GC=4096
// cls: B=16,S=256  -> 8 groups of 512,   C=128, rows/expert GC=1024

typedef __attribute__((ext_vector_type(16))) __bf16 v16bf;
typedef __attribute__((ext_vector_type(8)))  float  v8f;
typedef __attribute__((ext_vector_type(4)))  unsigned int v4u;
typedef __attribute__((ext_vector_type(8)))  int    v8i;
typedef __attribute__((ext_vector_type(4)))  int    v4i;

#if __has_builtin(__builtin_amdgcn_tensor_load_to_lds)
#define USE_TDM 1
#if __has_include(<hip/amd_detail/amd_gfx1250_TDM.h>)
#define TDM_6ARG 1   // therock/clang-23 lane: 6-arg builtin
#endif
#endif

union FragU {
    v16bf v;
    unsigned int w[8];
    uint4 q[2];
};

struct TokSlot { int erow; float wgt; };   // erow = e*GC + g*C + c, or -1

// float -> bf16 (round-to-nearest-even)
__device__ __forceinline__ unsigned short f2bf(float f) {
    union { float f; unsigned u; } c; c.f = f;
    unsigned r = c.u + 0x7FFFu + ((c.u >> 16) & 1u);
    return (unsigned short)(r >> 16);
}

// jax.nn.gelu default (approximate=True, tanh form)
__device__ __forceinline__ float gelu_tanh(float x) {
    float x3 = x * x * x;
    return 0.5f * x * (1.0f + tanhf(0.7978845608028654f * (x + 0.044715f * x3)));
}

// Even-K base for 16-bit A/B fragment element pair v (ISA 7.12.2, 16-bit A 16x32):
// lanes 0-15: VGPR0-3 -> K=0..7, VGPR4-7 -> K=16..23 ; lanes 16-31: +8
__device__ __forceinline__ int keven(int v, int lane) {
    return 2 * v + (v >= 4 ? 8 : 0) + (lane < 16 ? 0 : 8);
}

// Load one 16x32 (MxK) A fragment from LDS stored row-major [16][ld] as bf16 bits.
__device__ __forceinline__ v16bf load_a_lds(const unsigned short* base, int k0, int ld, int lane) {
    FragU f;
    int m = lane & 15;
#pragma unroll
    for (int v = 0; v < 8; ++v) {
        int k = k0 + keven(v, lane);                     // even, pair (k, k+1)
        f.w[v] = *(const unsigned int*)(base + m * ld + k);
    }
    return f.v;
}

// Load one 32x16 (KxN) B fragment from the packed weight image:
// tile is 512 bf16; lane's 16 elements are contiguous 32 bytes at lane*16.
__device__ __forceinline__ v16bf load_b_pack(const unsigned short* tile, int lane) {
    FragU f;
    const uint4* p = (const uint4*)(tile + lane * 16);
    f.q[0] = p[0];
    f.q[1] = p[1];
    return f.v;
}

// ---------------------------------------------------------------------------
// Kernel 0: fp32 -> bf16 bulk convert (for TDM-sourced activations)
// ---------------------------------------------------------------------------
__global__ __launch_bounds__(256) void cvt_bf16_kernel(const float* __restrict__ src,
                                                       unsigned short* __restrict__ dst,
                                                       int n) {
    int i = blockIdx.x * 256 + threadIdx.x;
    if (i < n) dst[i] = f2bf(src[i]);
}

// ---------------------------------------------------------------------------
// Kernel 1: pack fp32 weights [E][K=KT*32][N=NT*16] -> bf16 fragment-major tiles
// tile index t = (e*KT + kt)*NT + nt ; entry = lane*16 + el
// ---------------------------------------------------------------------------
__global__ __launch_bounds__(256) void pack_w_kernel(const float* __restrict__ w,
                                                     unsigned short* __restrict__ outp,
                                                     int KT, int NT, int ldN) {
    long t = blockIdx.x;
    int nt = (int)(t % NT);
    long r = t / NT;
    int kt = (int)(r % KT);
    int e  = (int)(r / KT);
    const float* src = w + (size_t)e * (size_t)(KT * 32) * (size_t)ldN;
    unsigned short* dst = outp + (size_t)t * 512;
    for (int idx = threadIdx.x; idx < 512; idx += 256) {
        int lane = idx >> 4;
        int el   = idx & 15;
        int v = el >> 1, odd = el & 1;
        int K = kt * 32 + keven(v, lane) + odd;
        int N = nt * 16 + (lane & 15);
        dst[idx] = f2bf(src[(size_t)K * ldN + N]);
    }
}

// ---------------------------------------------------------------------------
// Kernel 2: router — one block per group of S tokens.
// ---------------------------------------------------------------------------
__global__ __launch_bounds__(256) void router_kernel(const float* __restrict__ x,
                                                     const float* __restrict__ wr,
                                                     int S, int C, int GC,
                                                     int* __restrict__ assign_tok,
                                                     TokSlot* __restrict__ slots) {
    __shared__ float swr[D_MODEL * NUM_EXPERTS];   // 24 KB
    __shared__ float sv0[1024], sv1[1024];
    __shared__ int   si0[1024], si1[1024];
    const int g = blockIdx.x;

    for (int i = threadIdx.x; i < D_MODEL * NUM_EXPERTS; i += 256) swr[i] = wr[i];
    __syncthreads();

    for (int s = threadIdx.x; s < S; s += 256) {
        const float* xr = x + (size_t)(g * S + s) * D_MODEL;
        float acc[NUM_EXPERTS];
#pragma unroll
        for (int e = 0; e < NUM_EXPERTS; ++e) acc[e] = 0.f;
        for (int d = 0; d < D_MODEL; ++d) {
            float xv = xr[d];
#pragma unroll
            for (int e = 0; e < NUM_EXPERTS; ++e) acc[e] += xv * swr[d * NUM_EXPERTS + e];
        }
        float mx = acc[0];
#pragma unroll
        for (int e = 1; e < NUM_EXPERTS; ++e) mx = fmaxf(mx, acc[e]);
        float p[NUM_EXPERTS], sum = 0.f;
#pragma unroll
        for (int e = 0; e < NUM_EXPERTS; ++e) { p[e] = __expf(acc[e] - mx); sum += p[e]; }
        float inv = 1.0f / sum;
        int i0 = 0; float v0 = p[0];
#pragma unroll
        for (int e = 1; e < NUM_EXPERTS; ++e) if (p[e] > v0) { v0 = p[e]; i0 = e; }
        int i1 = -1; float v1 = -1.f;
#pragma unroll
        for (int e = 0; e < NUM_EXPERTS; ++e) if (e != i0 && p[e] > v1) { v1 = p[e]; i1 = e; }
        sv0[s] = v0 * inv; si0[s] = i0;
        sv1[s] = v1 * inv; si1[s] = i1;
    }
    __syncthreads();

    for (int i = threadIdx.x; i < NUM_EXPERTS * C; i += 256) {
        int e = i / C, c = i % C;
        assign_tok[(size_t)e * GC + g * C + c] = -1;
    }
    __syncthreads();

    // serial k-major capacity assignment (reference cumsum dispatcher)
    if (threadIdx.x == 0) {
        int cnt[NUM_EXPERTS];
#pragma unroll
        for (int e = 0; e < NUM_EXPERTS; ++e) cnt[e] = 0;
        for (int k = 0; k < 2; ++k) {
            for (int s = 0; s < S; ++s) {
                int   e  = (k == 0) ? si0[s] : si1[s];
                float wv = (k == 0) ? sv0[s] : sv1[s];
                int pos = cnt[e]++;
                TokSlot ts;
                if (pos < C) {
                    int row = g * C + pos;
                    assign_tok[(size_t)e * GC + row] = g * S + s;
                    ts.erow = e * GC + row;
                    ts.wgt  = wv;
                } else {
                    ts.erow = -1;
                    ts.wgt  = 0.f;
                }
                slots[(size_t)(g * S + s) * 2 + k] = ts;
            }
        }
    }
}

// ---------------------------------------------------------------------------
// TDM gather-mode descriptor issue: pull 16 rows (bf16, 768 wide) of xbf into
// LDS at lds_off. Row indices packed 16-bit in D# groups 2/3 (ISA 8.7).
// ---------------------------------------------------------------------------
#if defined(USE_TDM)
__device__ __forceinline__ void tdm_gather16(const unsigned short* xbf, int numTokens,
                                             unsigned lds_off, const int* toks) {
    unsigned long long ga = (unsigned long long)(size_t)xbf;
    v4u g0 = { 0x80000001u,                      // count=1, gather_mode=1, 16-bit idx
               lds_off,
               (unsigned)(ga & 0xFFFFFFFFu),
               (unsigned)((ga >> 32) & 0x01FFFFFFu) | 0x80000000u };  // type=2
    v8i g1 = { (int)0x00010000,                  // data_size=1 (2 bytes)
               (int)(((unsigned)D_MODEL & 0xFFFFu) << 16),      // tensor_dim0[15:0]
               (int)(((unsigned)numTokens & 0xFFFFu) << 16),    // dim0 hi=0 | tensor_dim1 lo
               (int)(((unsigned)D_MODEL & 0xFFFFu) << 16),      // dim1 hi=0 | tile_dim0=768
               (int)16,                          // tile_dim1 = #valid indices
               (int)D_MODEL,                     // tensor_dim0_stride lo32
               0, 0 };
    v4i g2 = { toks[0]  | (toks[1]  << 16), toks[2]  | (toks[3]  << 16),
               toks[4]  | (toks[5]  << 16), toks[6]  | (toks[7]  << 16) };
    v4i g3 = { toks[8]  | (toks[9]  << 16), toks[10] | (toks[11] << 16),
               toks[12] | (toks[13] << 16), toks[14] | (toks[15] << 16) };
#if defined(TDM_6ARG)
    __builtin_amdgcn_tensor_load_to_lds(g0, g1, g2, g3, (v8i)0, 0);
#else
    __builtin_amdgcn_tensor_load_to_lds(g0, g1, g2, g3, 0);
#endif
}
#endif

// ---------------------------------------------------------------------------
// Kernel 3: fused expert MLP via WMMA bf16. M=32 row tile (2 subtiles) so each
// B fragment feeds two WMMAs. block = (rowTile, expert), 256 threads = 8 waves.
// ---------------------------------------------------------------------------
__global__ __launch_bounds__(256) void moe_mlp_kernel(const unsigned short* __restrict__ xbf,
                                                      const int* __restrict__ assign_tok,
                                                      const unsigned short* __restrict__ w1p,
                                                      const unsigned short* __restrict__ w2p,
                                                      const float* __restrict__ b1,
                                                      const float* __restrict__ b2,
                                                      float* __restrict__ Y,
                                                      int GC, int numTokens) {
    __shared__ unsigned short ldsX[32 * D_MODEL];  // 48 KB: gathered A tile (bf16 row-major)
    __shared__ unsigned short ldsH[32 * 128];      //  8 KB: staged hidden chunk
    const int e = blockIdx.y;
    const int rowBase = blockIdx.x * 32;
    const int tid  = threadIdx.x;
    const int lane = tid & 31;
    const int wv   = tid >> 5;                     // 8 waves

#if defined(USE_TDM)
    if (wv == 0) {
        int toks[32];
#pragma unroll
        for (int i = 0; i < 32; ++i) {
            int t = assign_tok[(size_t)e * GC + rowBase + i];
            toks[i] = (t < 0) ? 0 : t;             // empty slot -> row 0 (never combined)
        }
        unsigned lds0 = (unsigned)(size_t)(&ldsX[0]);
        tdm_gather16(xbf, numTokens, lds0, toks);
        tdm_gather16(xbf, numTokens, lds0 + 16u * D_MODEL * 2u, toks + 16);
        __builtin_amdgcn_s_wait_tensorcnt(0);
    }
    __syncthreads();
#else
    __shared__ int ldsTok[32];
    if (tid < 32) ldsTok[tid] = assign_tok[(size_t)e * GC + rowBase + tid];
    __syncthreads();
    for (int i = tid; i < 32 * D_MODEL; i += 256) {
        int m = i / D_MODEL, d = i % D_MODEL;
        int tok = ldsTok[m];
        ldsX[i] = (tok >= 0) ? xbf[(size_t)tok * D_MODEL + d] : (unsigned short)0;
    }
    __syncthreads();
#endif

    const v8f vzero = {};
    v8f accY0[6], accY1[6];
#pragma unroll
    for (int t = 0; t < 6; ++t) { accY0[t] = vzero; accY1[t] = vzero; }

    const int m_off = (lane < 16) ? 0 : 8;         // D-matrix: M = r + m_off
    const int n_off = lane & 15;                   //           N = lane & 15

    for (int hc = 0; hc < 24; ++hc) {              // 24 chunks of 128 hidden cols
        // ---- phase 1: H chunk rows 0..31, this wave's 16 cols ----
        v8f accH0 = vzero, accH1 = vzero;
#pragma unroll 4
        for (int kt = 0; kt < 24; ++kt) {
            const unsigned short* bt =
                w1p + (((size_t)e * 24 + kt) * 192 + (size_t)(hc * 8 + wv)) * 512;
            __builtin_prefetch(bt + (size_t)192 * 512, 0, 1);  // next kt tile
            v16bf b  = load_b_pack(bt, lane);
            v16bf a0 = load_a_lds(ldsX,                kt * 32, D_MODEL, lane);
            v16bf a1 = load_a_lds(ldsX + 16 * D_MODEL, kt * 32, D_MODEL, lane);
            accH0 = __builtin_amdgcn_wmma_f32_16x16x32_bf16(false, a0, false, b,
                                                            (short)0, accH0, false, false);
            accH1 = __builtin_amdgcn_wmma_f32_16x16x32_bf16(false, a1, false, b,
                                                            (short)0, accH1, false, false);
        }
        {
            int hloc = wv * 16 + n_off;
            float bb = b1[(size_t)e * MLP_DIM + hc * 128 + hloc];
#pragma unroll
            for (int r = 0; r < 8; ++r) {
                ldsH[(r + m_off) * 128 + hloc]        = f2bf(gelu_tanh(accH0[r] + bb));
                ldsH[(16 + r + m_off) * 128 + hloc]   = f2bf(gelu_tanh(accH1[r] + bb));
            }
        }
        __syncthreads();

        // ---- phase 2: Y(32 x 96-per-wave) += H(32x128) @ W2 chunk ----
#pragma unroll
        for (int kt2 = 0; kt2 < 4; ++kt2) {
            v16bf a0 = load_a_lds(ldsH,            kt2 * 32, 128, lane);
            v16bf a1 = load_a_lds(ldsH + 16 * 128, kt2 * 32, 128, lane);
            int ktg = hc * 4 + kt2;
#pragma unroll
            for (int t = 0; t < 6; ++t) {
                const unsigned short* bt =
                    w2p + (((size_t)e * 96 + ktg) * 48 + (size_t)(wv * 6 + t)) * 512;
                v16bf b = load_b_pack(bt, lane);
                accY0[t] = __builtin_amdgcn_wmma_f32_16x16x32_bf16(false, a0, false, b,
                                                                   (short)0, accY0[t], false, false);
                accY1[t] = __builtin_amdgcn_wmma_f32_16x16x32_bf16(false, a1, false, b,
                                                                   (short)0, accY1[t], false, false);
            }
        }
        __syncthreads();
    }

    // write Y + b2 (fp32)
#pragma unroll
    for (int t = 0; t < 6; ++t) {
        int n = (wv * 6 + t) * 16 + n_off;
        float bb = b2[(size_t)e * D_MODEL + n];
#pragma unroll
        for (int r = 0; r < 8; ++r) {
            Y[((size_t)e * GC + rowBase + r + m_off) * D_MODEL + n]      = accY0[t][r] + bb;
            Y[((size_t)e * GC + rowBase + 16 + r + m_off) * D_MODEL + n] = accY1[t][r] + bb;
        }
    }
}

// ---------------------------------------------------------------------------
// Kernel 4: combine — per token, out = sum_k gate_k * Y[slot_k]
// Output concat layout: out[b, 0:1024]=det, out[b, 1024:1280]=cls.
// ---------------------------------------------------------------------------
__global__ __launch_bounds__(256) void combine_kernel(const TokSlot* __restrict__ slots,
                                                      const float* __restrict__ Y,
                                                      float* __restrict__ outp,
                                                      int tokPerBatch, int seqOff) {
    int f = blockIdx.x;
    int b = f / tokPerBatch;
    int s = f % tokPerBatch;
    float* o = outp + ((size_t)b * 1280 + seqOff + s) * D_MODEL;
    TokSlot s0 = slots[(size_t)f * 2 + 0];
    TokSlot s1 = slots[(size_t)f * 2 + 1];
    for (int d = threadIdx.x; d < D_MODEL; d += 256) {
        float v = 0.f;
        if (s0.erow >= 0) v += s0.wgt * Y[(size_t)s0.erow * D_MODEL + d];
        if (s1.erow >= 0) v += s1.wgt * Y[(size_t)s1.erow * D_MODEL + d];
        o[d] = v;
    }
}

// ---------------------------------------------------------------------------
// Launch
// ---------------------------------------------------------------------------
extern "C" void kernel_launch(void* const* d_in, const int* in_sizes, int n_in,
                              void* d_out, int out_size, void* d_ws, size_t ws_size,
                              hipStream_t stream) {
    (void)in_sizes; (void)n_in; (void)out_size; (void)ws_size;

    const float* x_det  = (const float*)d_in[0];   // [16,1024,768]
    const float* x_cls  = (const float*)d_in[1];   // [16,256,768]
    const float* wr_det = (const float*)d_in[2];   // [768,8]
    const float* wr_cls = (const float*)d_in[3];   // [768,8]
    const float* w1     = (const float*)d_in[4];   // [8,768,3072]
    const float* b1     = (const float*)d_in[5];   // [8,3072]
    const float* w2     = (const float*)d_in[6];   // [8,3072,768]
    const float* b2     = (const float*)d_in[7];   // [8,768]
    float* outp = (float*)d_out;                   // [16,1280,768]

    char* ws = (char*)d_ws;
    size_t off = 0;
    auto carve = [&](size_t bytes) -> char* {
        char* p = ws + off;
        off += (bytes + 255) & ~(size_t)255;
        return p;
    };
    const size_t W1_TILES = (size_t)NUM_EXPERTS * 24 * 192;  // 36864
    const size_t W2_TILES = (size_t)NUM_EXPERTS * 96 * 48;   // 36864
    unsigned short* w1p = (unsigned short*)carve(W1_TILES * 512 * 2);
    unsigned short* w2p = (unsigned short*)carve(W2_TILES * 512 * 2);

    const int GC_DET = 16 * 256;   // 4096 rows/expert
    const int GC_CLS = 8 * 128;    // 1024 rows/expert
    const int NTOK_DET = 16 * 1024;
    const int NTOK_CLS = 16 * 256;
    int*     assign_det = (int*)carve((size_t)NUM_EXPERTS * GC_DET * 4);
    int*     assign_cls = (int*)carve((size_t)NUM_EXPERTS * GC_CLS * 4);
    TokSlot* slots_det  = (TokSlot*)carve((size_t)NTOK_DET * 2 * sizeof(TokSlot));
    TokSlot* slots_cls  = (TokSlot*)carve((size_t)NTOK_CLS * 2 * sizeof(TokSlot));
    float*   Y_det      = (float*)carve((size_t)NUM_EXPERTS * GC_DET * D_MODEL * 4);
    float*   Y_cls      = (float*)carve((size_t)NUM_EXPERTS * GC_CLS * D_MODEL * 4);
    unsigned short* xbf_det = (unsigned short*)carve((size_t)NTOK_DET * D_MODEL * 2);
    unsigned short* xbf_cls = (unsigned short*)carve((size_t)NTOK_CLS * D_MODEL * 2);

    // 0) activations to bf16 (TDM source)
    {
        int nd = NTOK_DET * D_MODEL, nc = NTOK_CLS * D_MODEL;
        cvt_bf16_kernel<<<(nd + 255) / 256, 256, 0, stream>>>(x_det, xbf_det, nd);
        cvt_bf16_kernel<<<(nc + 255) / 256, 256, 0, stream>>>(x_cls, xbf_cls, nc);
    }

    // 1) pack weights to bf16 fragment-major
    pack_w_kernel<<<dim3((unsigned)W1_TILES), 256, 0, stream>>>(w1, w1p, 24, 192, MLP_DIM);
    pack_w_kernel<<<dim3((unsigned)W2_TILES), 256, 0, stream>>>(w2, w2p, 96, 48, D_MODEL);

    // 2) routing
    router_kernel<<<16, 256, 0, stream>>>(x_det, wr_det, 1024, 256, GC_DET, assign_det, slots_det);
    router_kernel<<<8,  256, 0, stream>>>(x_cls, wr_cls, 512,  128, GC_CLS, assign_cls, slots_cls);

    // 3) fused expert MLP (TDM gather + GEMM1 + gelu + GEMM2), M=32 tiles
    moe_mlp_kernel<<<dim3(GC_DET / 32, NUM_EXPERTS), 256, 0, stream>>>(
        xbf_det, assign_det, w1p, w2p, b1, b2, Y_det, GC_DET, NTOK_DET);
    moe_mlp_kernel<<<dim3(GC_CLS / 32, NUM_EXPERTS), 256, 0, stream>>>(
        xbf_cls, assign_cls, w1p, w2p, b1, b2, Y_cls, GC_CLS, NTOK_CLS);

    // 4) combine into concat output
    combine_kernel<<<NTOK_DET, 256, 0, stream>>>(slots_det, Y_det, outp, 1024, 0);
    combine_kernel<<<NTOK_CLS, 256, 0, stream>>>(slots_cls, Y_cls, outp, 256, 1024);
}
